// DeltaNet_65034394796870
// MI455X (gfx1250) — matrix operand order repaired and checked by reading, not compiled, not run on hardware
//
#include <hip/hip_runtime.h>
#include <hip/hip_bf16.h>
#include <stdint.h>

// ---------------------------------------------------------------------------
// Problem constants (from reference)
// ---------------------------------------------------------------------------
static constexpr int BB  = 2;
static constexpr int LL  = 4096;
static constexpr int DD  = 1024;   // model dim
static constexpr int HH  = 16;     // heads
static constexpr int HDm = 64;     // head dim
static constexpr int KC  = 4;      // conv kernel
static constexpr int NT  = BB * LL;          // 8192 token rows
static constexpr float EPSF = 1e-5f;

typedef __attribute__((ext_vector_type(16))) __bf16 v16bf;
typedef __attribute__((ext_vector_type(8)))  float  v8f;
typedef int v4i __attribute__((vector_size(16)));   // matches builtin prototype

// ---- gfx1250 async global<->LDS copy availability (probe via __has_builtin) --
#if defined(__has_builtin)
#  if __has_builtin(__builtin_amdgcn_global_load_async_to_lds_b128)
#    define HAVE_ASYNC_LD 1
#  else
#    define HAVE_ASYNC_LD 0
#  endif
#else
#  define HAVE_ASYNC_LD 0
#endif

#define GAS __attribute__((address_space(1)))
#define LAS __attribute__((address_space(3)))
#define GASP(p) ((GAS v4i*)(p))
#define LASP(p) ((LAS v4i*)(p))

__device__ __forceinline__ void wait_async0() {
    asm volatile("s_wait_asynccnt 0x0" ::: "memory");
}

// ---------------------------------------------------------------------------
// fp32 -> bf16 (RNE) helpers
// ---------------------------------------------------------------------------
__device__ __forceinline__ unsigned short f32_to_bf16(float f) {
    unsigned u = __float_as_uint(f);
    unsigned r = 0x7FFFu + ((u >> 16) & 1u);
    return (unsigned short)((u + r) >> 16);
}

__global__ __launch_bounds__(256) void convert_bf16_kernel(const float* __restrict__ in,
                                                           unsigned short* __restrict__ out,
                                                           int n) {
    int i0 = (blockIdx.x * 256 + threadIdx.x) * 4;
    if (i0 + 3 < n) {
        float4 v = *(const float4*)(in + i0);
        out[i0 + 0] = f32_to_bf16(v.x);
        out[i0 + 1] = f32_to_bf16(v.y);
        out[i0 + 2] = f32_to_bf16(v.z);
        out[i0 + 3] = f32_to_bf16(v.w);
    } else {
        for (int i = i0; i < n; ++i) out[i] = f32_to_bf16(in[i]);
    }
}

// ---------------------------------------------------------------------------
// bf16 WMMA GEMM:  C[Nrows,1024] (f32) = A[Nrows,1024] (bf16) @ W[1024,1024] (bf16)
// Block tile: 256 (M) x 64 (N); 8 wave32s, each computing a 32x64 strip as
// 8x v_wmma_f32_16x16x32_bf16 per 32-wide K step (2 A frags x 4 B frags).
// Double-buffered: tile i+1 async-copied global->LDS behind tile i's WMMAs;
// one barrier per K step.
// ---------------------------------------------------------------------------
static constexpr int GK = 1024;  // inner dim
static constexpr int GN = 1024;  // output dim
static constexpr int BSTR = 36;  // LDS row stride for transposed B tile (even)

__global__ __launch_bounds__(256) void gemm_bf16_wmma_kernel(const unsigned short* __restrict__ A,
                                                             const unsigned short* __restrict__ Bm,
                                                             float* __restrict__ C,
                                                             int Nrows) {
    __shared__ unsigned short As[2][256 * 32];   // 2 x 16 KB
    __shared__ unsigned short Bs[2][64 * BSTR];  // transposed: Bs[p][n*BSTR + k]

    const int t    = threadIdx.x;
    const int wave = t >> 5;          // 0..7
    const int lane = t & 31;
    const int row0 = blockIdx.y * 256;
    const int col0 = blockIdx.x * 64;

    v8f z = {};
    v8f acc[8];                       // [frag 0..1][ntile 0..3]
#pragma unroll
    for (int i = 0; i < 8; ++i) acc[i] = z;

    // Fragment address precompute (contiguous K per lane half, same order for A and B).
    const int mloc   = lane & 15;
    const int khalfA = (lane >> 4) * 16;   // A: lanes 0-15 -> K 0..15, lanes 16-31 -> K 16..31
    const int khalfB = (lane >> 4) * 16;   // B: lanes 0-15 -> K 0..15, lanes 16-31 -> K 16..31

    // B staging coords for this thread
    const int bi  = t & 31;           // k within tile
    const int bn0 = (t >> 5) * 8;     // 8 consecutive n

    // ---- prologue: stage tile 0 into buffer 0 ----
    {
        const unsigned short* g = A + (size_t)(row0 + t) * GK;
        unsigned short* s = &As[0][t * 32];
#if HAVE_ASYNC_LD
        __builtin_amdgcn_global_load_async_to_lds_b128(GASP(g + 0),  LASP(s + 0),  0, 0);
        __builtin_amdgcn_global_load_async_to_lds_b128(GASP(g + 8),  LASP(s + 8),  0, 0);
        __builtin_amdgcn_global_load_async_to_lds_b128(GASP(g + 16), LASP(s + 16), 0, 0);
        __builtin_amdgcn_global_load_async_to_lds_b128(GASP(g + 24), LASP(s + 24), 0, 0);
#else
        const uint4* gv = (const uint4*)g;
        uint4* sv = (uint4*)s;
        sv[0] = gv[0]; sv[1] = gv[1]; sv[2] = gv[2]; sv[3] = gv[3];
#endif
        uint4 gB = *(const uint4*)(Bm + (size_t)bi * GN + col0 + bn0);
        const unsigned short* gs = (const unsigned short*)&gB;
#pragma unroll
        for (int q = 0; q < 8; ++q) Bs[0][(bn0 + q) * BSTR + bi] = gs[q];
    }

    constexpr int NIT = GK / 32;      // 32 K-steps
    for (int it = 0; it < NIT; ++it) {
        const int p = it & 1;
        const bool hasNext = (it + 1) < NIT;

#if HAVE_ASYNC_LD
        wait_async0();                // tile p's async A copies arrived
#endif
        __syncthreads();              // tile p visible; buffer p^1 free of readers

        // ---- issue next tile's loads (drain behind this tile's WMMAs) ----
        uint4 bnext = {0, 0, 0, 0};
        if (hasNext) {
            const int kkn = (it + 1) * 32;
            const unsigned short* g = A + (size_t)(row0 + t) * GK + kkn;
            unsigned short* s = &As[p ^ 1][t * 32];
#if HAVE_ASYNC_LD
            __builtin_amdgcn_global_load_async_to_lds_b128(GASP(g + 0),  LASP(s + 0),  0, 0);
            __builtin_amdgcn_global_load_async_to_lds_b128(GASP(g + 8),  LASP(s + 8),  0, 0);
            __builtin_amdgcn_global_load_async_to_lds_b128(GASP(g + 16), LASP(s + 16), 0, 0);
            __builtin_amdgcn_global_load_async_to_lds_b128(GASP(g + 24), LASP(s + 24), 0, 0);
#else
            const uint4* gv = (const uint4*)g;
            uint4* sv = (uint4*)s;
            sv[0] = gv[0]; sv[1] = gv[1]; sv[2] = gv[2]; sv[3] = gv[3];
#endif
            bnext = *(const uint4*)(Bm + (size_t)(kkn + bi) * GN + col0 + bn0);
        }

        // ---- build two A fragments for this wave's 32-row strip ----
        union { v16bf v; unsigned int u[8]; } fa[2];
#pragma unroll
        for (int f = 0; f < 2; ++f) {
            const unsigned int* ap =
                (const unsigned int*)(&As[p][(wave * 32 + f * 16 + mloc) * 32]);
#pragma unroll
            for (int i = 0; i < 4; ++i) fa[f].u[i] = ap[(khalfA + 2 * i) >> 1];
#pragma unroll
            for (int i = 4; i < 8; ++i) fa[f].u[i] = ap[(8 + khalfA + 2 * (i - 4)) >> 1];
        }
        // ---- 4 N-tiles, each feeding both A fragments ----
#pragma unroll
        for (int nt = 0; nt < 4; ++nt) {
            union { v16bf v; unsigned int u[8]; } fb;
            const unsigned int* bp =
                (const unsigned int*)(&Bs[p][(nt * 16 + mloc) * BSTR + khalfB]);
#pragma unroll
            for (int i = 0; i < 8; ++i) fb.u[i] = bp[i];

            acc[nt] = __builtin_amdgcn_wmma_f32_16x16x32_bf16(
                false, fa[0].v, false, fb.v, (short)0, acc[nt], false, false);
            acc[4 + nt] = __builtin_amdgcn_wmma_f32_16x16x32_bf16(
                false, fa[1].v, false, fb.v, (short)0, acc[4 + nt], false, false);
        }

        // ---- scatter next B tile (transposed) into buffer p^1 ----
        if (hasNext) {
            const unsigned short* gs = (const unsigned short*)&bnext;
#pragma unroll
            for (int q = 0; q < 8; ++q) Bs[p ^ 1][(bn0 + q) * BSTR + bi] = gs[q];
        }
    }

    // ---- epilogue: C/D layout -> VGPR r holds M = r (lanes 0-15) / r+8 (16-31) ----
    const int mAdd = (lane >> 4) * 8;
#pragma unroll
    for (int f = 0; f < 2; ++f) {
#pragma unroll
        for (int nt = 0; nt < 4; ++nt) {
            int col = col0 + nt * 16 + mloc;
#pragma unroll
            for (int r = 0; r < 8; ++r) {
                int row = row0 + wave * 32 + f * 16 + r + mAdd;
                C[(size_t)row * GN + col] = acc[f * 4 + nt][r];
            }
        }
    }
}

// ---------------------------------------------------------------------------
// beta = sigmoid(x @ Wb)   x:[NT,D] f32, Wb:[D,16] f32 -> beta:[NT,16] f32
// ---------------------------------------------------------------------------
__global__ __launch_bounds__(256) void beta_kernel(const float* __restrict__ x,
                                                   const float* __restrict__ Wb,
                                                   float* __restrict__ beta) {
    int idx = blockIdx.x * 256 + threadIdx.x;   // NT*H outputs
    int n = idx >> 4, h = idx & 15;
    const float* xr = x + (size_t)n * DD;
    float acc = 0.f;
#pragma unroll 8
    for (int k = 0; k < DD; ++k) acc = fmaf(xr[k], Wb[k * HH + h], acc);
    beta[idx] = 1.f / (1.f + __expf(-acc));
}

// ---------------------------------------------------------------------------
// causal depthwise conv (K=4) + SiLU (+ optional per-head L2 norm)
// ---------------------------------------------------------------------------
template <bool NORM>
__global__ __launch_bounds__(256) void conv_act_kernel(const float* __restrict__ in,
                                                       const float* __restrict__ cw,
                                                       float* __restrict__ out) {
    int bl = blockIdx.x;
    int b = bl / LL, l = bl % LL;
    int t = threadIdx.x;
    int d0 = t * 4;

    float y[4];
#pragma unroll
    for (int c = 0; c < 4; ++c) {
        int d = d0 + c;
        float acc = 0.f;
#pragma unroll
        for (int j = 0; j < KC; ++j) {
            int lp = l + j - (KC - 1);
            if (lp >= 0)
                acc = fmaf(in[((size_t)b * LL + lp) * DD + d], cw[d * KC + j], acc);
        }
        y[c] = acc * (1.f / (1.f + __expf(-acc)));   // SiLU
    }

    if (NORM) {
        float ss = y[0] * y[0] + y[1] * y[1] + y[2] * y[2] + y[3] * y[3];
        ss += __shfl_xor(ss, 1, 32);
        ss += __shfl_xor(ss, 2, 32);
        ss += __shfl_xor(ss, 4, 32);
        ss += __shfl_xor(ss, 8, 32);   // 16 lanes == one head (64 channels)
        float inv = rsqrtf(ss);
#pragma unroll
        for (int c = 0; c < 4; ++c) y[c] *= inv;
    }
    float4 o4 = {y[0], y[1], y[2], y[3]};
    *(float4*)(out + (size_t)bl * DD + d0) = o4;
}

// ---------------------------------------------------------------------------
// Delta-rule scan. One block per (b,h). State S[64][64] f32 in registers:
// thread t -> column j = t>>2, i-range = (t&3)*16 .. +15. Columns independent
// given shared k/v/q/beta -> no per-step barriers. Double-buffered chunks:
// chunk c+1 async-copied global->LDS behind chunk c's 64 recurrence steps.
// ---------------------------------------------------------------------------
static constexpr int CHUNK = 64;

__global__ __launch_bounds__(256) void scan_kernel(const float* __restrict__ Q,
                                                   const float* __restrict__ Kt,
                                                   const float* __restrict__ V,
                                                   const float* __restrict__ beta,
                                                   float* __restrict__ O) {
    __shared__ float lq[2][CHUNK * HDm];
    __shared__ float lk[2][CHUNK * HDm];
    __shared__ float lv[2][CHUNK * HDm];
    __shared__ float lo[CHUNK * HDm];
    __shared__ float lb[2][CHUNK];

    const int bid = blockIdx.x;
    const int b = bid / HH, h = bid % HH;
    const int t = threadIdx.x;
    const int j = t >> 2;            // output column 0..63
    const int sub = t & 3;           // i-subrange 0..3
    const int ib = sub * 16;

    float4 S0 = {0,0,0,0}, S1 = S0, S2 = S0, S3 = S0;   // S[ib..ib+15][j]

    // ---- prologue: stage chunk 0 into buffer 0 ----
#pragma unroll
    for (int r = 0; r < 4; ++r) {
        int f = r * 256 + t;
        int lr = f >> 4;
        int c4 = (f & 15) * 4;
        size_t g = ((size_t)b * LL + lr) * DD + h * HDm + c4;
        int lofs = lr * HDm + c4;
#if HAVE_ASYNC_LD
        __builtin_amdgcn_global_load_async_to_lds_b128(GASP(Q + g),  LASP(&lq[0][lofs]), 0, 0);
        __builtin_amdgcn_global_load_async_to_lds_b128(GASP(Kt + g), LASP(&lk[0][lofs]), 0, 0);
        __builtin_amdgcn_global_load_async_to_lds_b128(GASP(V + g),  LASP(&lv[0][lofs]), 0, 0);
#else
        *(float4*)(&lq[0][lofs]) = *(const float4*)(Q + g);
        *(float4*)(&lk[0][lofs]) = *(const float4*)(Kt + g);
        *(float4*)(&lv[0][lofs]) = *(const float4*)(V + g);
#endif
    }
    if (t < CHUNK) lb[0][t] = beta[((size_t)b * LL + t) * HH + h];

    constexpr int NC = LL / CHUNK;   // 64 chunks
    for (int ci = 0; ci < NC; ++ci) {
        const int p = ci & 1;
        const bool hasNext = (ci + 1) < NC;

#if HAVE_ASYNC_LD
        wait_async0();               // chunk ci's async copies arrived
#endif
        __syncthreads();             // visible to all; buffer p^1 free of readers

        // ---- issue next chunk's loads (drain behind this chunk's compute) ----
        if (hasNext) {
            int c0n = (ci + 1) * CHUNK;
#pragma unroll
            for (int r = 0; r < 4; ++r) {
                int f = r * 256 + t;
                int lr = f >> 4;
                int c4 = (f & 15) * 4;
                size_t g = ((size_t)b * LL + c0n + lr) * DD + h * HDm + c4;
                int lofs = lr * HDm + c4;
#if HAVE_ASYNC_LD
                __builtin_amdgcn_global_load_async_to_lds_b128(GASP(Q + g),  LASP(&lq[p ^ 1][lofs]), 0, 0);
                __builtin_amdgcn_global_load_async_to_lds_b128(GASP(Kt + g), LASP(&lk[p ^ 1][lofs]), 0, 0);
                __builtin_amdgcn_global_load_async_to_lds_b128(GASP(V + g),  LASP(&lv[p ^ 1][lofs]), 0, 0);
#else
                *(float4*)(&lq[p ^ 1][lofs]) = *(const float4*)(Q + g);
                *(float4*)(&lk[p ^ 1][lofs]) = *(const float4*)(Kt + g);
                *(float4*)(&lv[p ^ 1][lofs]) = *(const float4*)(V + g);
#endif
            }
            if (t < CHUNK) lb[p ^ 1][t] = beta[((size_t)b * LL + (ci + 1) * CHUNK + t) * HH + h];
        }

        // ---- 64 recurrence steps from buffer p ----
        const float* lqp = lq[p];
        const float* lkp = lk[p];
        const float* lvp = lv[p];
        const float* lbp = lb[p];
        for (int s = 0; s < CHUNK; ++s) {
            const float4* kp = (const float4*)(lkp + s * HDm + ib);
            float4 k0 = kp[0], k1 = kp[1], k2 = kp[2], k3 = kp[3];

            float pv = k0.x*S0.x + k0.y*S0.y + k0.z*S0.z + k0.w*S0.w
                     + k1.x*S1.x + k1.y*S1.y + k1.z*S1.z + k1.w*S1.w
                     + k2.x*S2.x + k2.y*S2.y + k2.z*S2.z + k2.w*S2.w
                     + k3.x*S3.x + k3.y*S3.y + k3.z*S3.z + k3.w*S3.w;
            pv += __shfl_xor(pv, 1, 32);
            pv += __shfl_xor(pv, 2, 32);         // 4-lane group per column

            float delta = (lvp[s * HDm + j] - pv) * lbp[s];

            S0.x = fmaf(k0.x, delta, S0.x); S0.y = fmaf(k0.y, delta, S0.y);
            S0.z = fmaf(k0.z, delta, S0.z); S0.w = fmaf(k0.w, delta, S0.w);
            S1.x = fmaf(k1.x, delta, S1.x); S1.y = fmaf(k1.y, delta, S1.y);
            S1.z = fmaf(k1.z, delta, S1.z); S1.w = fmaf(k1.w, delta, S1.w);
            S2.x = fmaf(k2.x, delta, S2.x); S2.y = fmaf(k2.y, delta, S2.y);
            S2.z = fmaf(k2.z, delta, S2.z); S2.w = fmaf(k2.w, delta, S2.w);
            S3.x = fmaf(k3.x, delta, S3.x); S3.y = fmaf(k3.y, delta, S3.y);
            S3.z = fmaf(k3.z, delta, S3.z); S3.w = fmaf(k3.w, delta, S3.w);

            const float4* qp = (const float4*)(lqp + s * HDm + ib);
            float4 q0 = qp[0], q1 = qp[1], q2 = qp[2], q3 = qp[3];
            float o = q0.x*S0.x + q0.y*S0.y + q0.z*S0.z + q0.w*S0.w
                    + q1.x*S1.x + q1.y*S1.y + q1.z*S1.z + q1.w*S1.w
                    + q2.x*S2.x + q2.y*S2.y + q2.z*S2.z + q2.w*S2.w
                    + q3.x*S3.x + q3.y*S3.y + q3.z*S3.z + q3.w*S3.w;
            o += __shfl_xor(o, 1, 32);
            o += __shfl_xor(o, 2, 32);
            if (sub == 0) lo[s * HDm + j] = o;
        }
        __syncthreads();             // lo complete

        // ---- flush output chunk (regular stores; drain behind next compute) ----
#pragma unroll
        for (int r = 0; r < 4; ++r) {
            int f = r * 256 + t;
            int lr = f >> 4;
            int c4 = (f & 15) * 4;
            size_t g = ((size_t)b * LL + ci * CHUNK + lr) * DD + h * HDm + c4;
            *(float4*)(O + g) = *(const float4*)(lo + lr * HDm + c4);
        }
    }
}

// ---------------------------------------------------------------------------
// per-head RMSNorm -> bf16 (feeds the final WMMA GEMM)
// ---------------------------------------------------------------------------
__global__ __launch_bounds__(256) void rmsnorm_bf16_kernel(const float* __restrict__ O,
                                                           const float* __restrict__ w,
                                                           unsigned short* __restrict__ Ob) {
    int bl = blockIdx.x;
    int t = threadIdx.x;
    int d0 = t * 4;
    float4 v = *(const float4*)(O + (size_t)bl * DD + d0);
    float ss = v.x*v.x + v.y*v.y + v.z*v.z + v.w*v.w;
    ss += __shfl_xor(ss, 1, 32);
    ss += __shfl_xor(ss, 2, 32);
    ss += __shfl_xor(ss, 4, 32);
    ss += __shfl_xor(ss, 8, 32);
    float rms = rsqrtf(ss * (1.0f / HDm) + EPSF);
    int i0 = d0 & (HDm - 1);
    Ob[(size_t)bl * DD + d0 + 0] = f32_to_bf16(v.x * rms * w[i0 + 0]);
    Ob[(size_t)bl * DD + d0 + 1] = f32_to_bf16(v.y * rms * w[i0 + 1]);
    Ob[(size_t)bl * DD + d0 + 2] = f32_to_bf16(v.z * rms * w[i0 + 2]);
    Ob[(size_t)bl * DD + d0 + 3] = f32_to_bf16(v.w * rms * w[i0 + 3]);
}

// ---------------------------------------------------------------------------
// host launcher
// ---------------------------------------------------------------------------
extern "C" void kernel_launch(void* const* d_in, const int* in_sizes, int n_in,
                              void* d_out, int out_size, void* d_ws, size_t ws_size,
                              hipStream_t stream) {
    (void)in_sizes; (void)n_in; (void)out_size; (void)ws_size;

    const float* x   = (const float*)d_in[0];
    const float* Wq  = (const float*)d_in[1];
    const float* Wk  = (const float*)d_in[2];
    const float* Wv  = (const float*)d_in[3];
    const float* Wb  = (const float*)d_in[4];
    const float* cq  = (const float*)d_in[5];
    const float* ck  = (const float*)d_in[6];
    const float* cv  = (const float*)d_in[7];
    const float* onw = (const float*)d_in[8];
    const float* Wo  = (const float*)d_in[9];
    float* out = (float*)d_out;

    // workspace carve-up
    char* base = (char*)d_ws;
    size_t off = 0;
    auto alloc = [&](size_t bytes) -> char* {
        char* p = base + off;
        off = (off + bytes + 255) & ~(size_t)255;
        return p;
    };
    unsigned short* xb  = (unsigned short*)alloc((size_t)NT * DD * 2);
    unsigned short* wqb = (unsigned short*)alloc((size_t)DD * DD * 2);
    unsigned short* wkb = (unsigned short*)alloc((size_t)DD * DD * 2);
    unsigned short* wvb = (unsigned short*)alloc((size_t)DD * DD * 2);
    unsigned short* wob = (unsigned short*)alloc((size_t)DD * DD * 2);
    float* Qp  = (float*)alloc((size_t)NT * DD * 4);
    float* Kp  = (float*)alloc((size_t)NT * DD * 4);
    float* Vp  = (float*)alloc((size_t)NT * DD * 4);
    float* Qa  = (float*)alloc((size_t)NT * DD * 4);
    float* Ka  = (float*)alloc((size_t)NT * DD * 4);
    float* Va  = (float*)alloc((size_t)NT * DD * 4);
    float* bet = (float*)alloc((size_t)NT * HH * 4);
    float* Osc = (float*)alloc((size_t)NT * DD * 4);
    unsigned short* Ob = (unsigned short*)alloc((size_t)NT * DD * 2);

    const int nx = NT * DD;       // 8,388,608
    const int nw = DD * DD;       // 1,048,576

    // 1) bf16 copies of activations + weights
    convert_bf16_kernel<<<nx / 1024, 256, 0, stream>>>(x,  xb,  nx);
    convert_bf16_kernel<<<nw / 1024, 256, 0, stream>>>(Wq, wqb, nw);
    convert_bf16_kernel<<<nw / 1024, 256, 0, stream>>>(Wk, wkb, nw);
    convert_bf16_kernel<<<nw / 1024, 256, 0, stream>>>(Wv, wvb, nw);
    convert_bf16_kernel<<<nw / 1024, 256, 0, stream>>>(Wo, wob, nw);

    // 2) q/k/v projections via bf16 WMMA (f32 accumulate)
    dim3 ggrid(GN / 64, NT / 256);
    gemm_bf16_wmma_kernel<<<ggrid, 256, 0, stream>>>(xb, wqb, Qp, NT);
    gemm_bf16_wmma_kernel<<<ggrid, 256, 0, stream>>>(xb, wkb, Kp, NT);
    gemm_bf16_wmma_kernel<<<ggrid, 256, 0, stream>>>(xb, wvb, Vp, NT);

    // 3) beta gate (f32, tiny)
    beta_kernel<<<(NT * HH) / 256, 256, 0, stream>>>(x, Wb, bet);

    // 4) conv + SiLU (+ L2 norm for q,k)
    conv_act_kernel<true ><<<NT, 256, 0, stream>>>(Qp, cq, Qa);
    conv_act_kernel<true ><<<NT, 256, 0, stream>>>(Kp, ck, Ka);
    conv_act_kernel<false><<<NT, 256, 0, stream>>>(Vp, cv, Va);

    // 5) delta-rule scan: one block per (b,h)
    scan_kernel<<<BB * HH, 256, 0, stream>>>(Qa, Ka, Va, bet, Osc);

    // 6) per-head RMSNorm -> bf16
    rmsnorm_bf16_kernel<<<NT, 256, 0, stream>>>(Osc, onw, Ob);

    // 7) output projection via bf16 WMMA, f32 result straight into d_out
    gemm_bf16_wmma_kernel<<<ggrid, 256, 0, stream>>>(Ob, wob, out, NT);
}
